// QuantumFeedForwardLayer_65481071402365
// MI455X (gfx1250) — compile-verified, matrix operand order checked by reading, not run on hardware
//
#include <hip/hip_runtime.h>

typedef __attribute__((ext_vector_type(16))) _Float16 v16h;
typedef __attribute__((ext_vector_type(8)))  _Float16 v8h;
typedef __attribute__((ext_vector_type(4)))  _Float16 v4h;
typedef __attribute__((ext_vector_type(8)))  float    v8f;
typedef __attribute__((ext_vector_type(4)))  float    v4f;
typedef __attribute__((ext_vector_type(4)))  int      v4i;

#define EMBED 2048
#define FFN   2048
#define NTOK  (8 * 4096)

#define BM 64
#define BN 256
#define BK 64    // K-depth per barrier window (2 wmma k-steps)
#define LDH 72   // halves per LDS row (144B stride = 36 dwords, conflict-free, 16B aligned)

// ---- gfx1250 async global->LDS path (guarded; falls back to plain loads) ----
#if defined(__has_builtin)
#if __has_builtin(__builtin_amdgcn_global_load_async_to_lds_b128)
#define HAVE_ASYNC_LDS 1
#endif
#endif
#ifndef HAVE_ASYNC_LDS
#define HAVE_ASYNC_LDS 0
#endif

// builtin signature (probe-confirmed): (int4 AS1* src, int4 AS3* dst, imm off, imm cpol)
typedef __attribute__((address_space(1))) v4i as1_v4i;
typedef __attribute__((address_space(3))) v4i as3_v4i;

__device__ __forceinline__ void wait_async_zero() {
#if __has_builtin(__builtin_amdgcn_s_wait_asynccnt)
  __builtin_amdgcn_s_wait_asynccnt(0);
#else
  asm volatile("s_wait_asynccnt 0x0" ::: "memory");
#endif
}

__device__ __forceinline__ v16h load_frag(const _Float16* p) {
  // wave32 16x32 f16 operand tile, row-major [16][LDH] in LDS:
  // lane L -> row L%16, k chunks {b..b+7} and {b+16..b+23}, b=(L/16)*8
  union { v16h v; v8h h[2]; } u;
  u.h[0] = *(const v8h*)p;
  u.h[1] = *(const v8h*)(p + 16);
  return u.v;
}

// ---------------- prep kernels ----------------

__global__ __launch_bounds__(256) void cvt_w2_f16(const float* __restrict__ W2,
                                                  _Float16* __restrict__ W2h, int n4) {
  int i = blockIdx.x * blockDim.x + threadIdx.x;
  if (i < n4) {
    v4f v = ((const v4f*)W2)[i];
    v4h h;
    h.x = (_Float16)v.x; h.y = (_Float16)v.y;
    h.z = (_Float16)v.z; h.w = (_Float16)v.w;
    ((v4h*)W2h)[i] = h;
  }
}

__global__ __launch_bounds__(256) void qfeat_kernel(const float* __restrict__ x,
                                                    float* __restrict__ q4, int ntok) {
  int t = blockIdx.x * blockDim.x + threadIdx.x;
  if (t < ntok) {
    v4f xv = *(const v4f*)(x + (size_t)t * EMBED);
    float c0 = cosf(xv.x), c1 = cosf(xv.y), c2 = cosf(xv.z), c3 = cosf(xv.w);
    v4f q; q.x = c0; q.y = c0 * c1; q.z = c2; q.w = c2 * c3;
    ((v4f*)q4)[t] = q;
  }
}

__global__ __launch_bounds__(256) void pack_a4(const float* __restrict__ W1,
                                               float* __restrict__ a4, int ffn) {
  int g = blockIdx.x * blockDim.x + threadIdx.x;
  if (g < ffn) ((v4f*)a4)[g] = *(const v4f*)(W1 + (size_t)g * EMBED);
}

// ------- fused rank4 -> relu -> GEMM, f16 WMMA w/ f32 acc, double-buffered LDS -------

__global__ __launch_bounds__(256)
void ffn_wmma_kernel(const float* __restrict__ qfeat,    // [NTOK] float4
                     const float* __restrict__ a4,       // [FFN]  float4 (W1[:,0:4])
                     const float* __restrict__ b1,       // [FFN]
                     const _Float16* __restrict__ W2h,   // [EMBED][FFN] f16
                     const float* __restrict__ b2,       // [EMBED]
                     float* __restrict__ out)            // [NTOK][EMBED]
{
  __shared__ __align__(16) v4f      qs[BM];
  __shared__ __align__(16) _Float16 Hs[2][BM][LDH];
  __shared__ __align__(16) _Float16 Bs[2][BN][LDH];

  const int t    = threadIdx.x;
  const int lane = t & 31;
  const int wave = t >> 5;
  const int wm   = (wave & 1) * 32;        // wave M offset: 0 / 32
  const int wn   = (wave >> 1) * 64;       // wave N offset: 0 / 64 / 128 / 192

  const int m_base = blockIdx.y * BM;      // token base
  const int n_base = blockIdx.x * BN;      // embed-column base

  // fragment addressing
  const int frag_row = lane & 15;
  const int frag_k   = (lane >> 4) * 8;

  // H staging: thread t -> row hm = t/4, 16-half chunk hg0 = (t%4)*16
  const int hm  = t >> 2;
  const int hg0 = (t & 3) * 16;

  // B staging: thread t -> one row (128B) of the 256x64 W2h tile
  const _Float16* brow = W2h + (size_t)(n_base + t) * FFN;

  if (t < BM) qs[t] = ((const v4f*)qfeat)[m_base + t];
  __syncthreads();  // qs visible before any H staging

  v8f acc[2][4] = {};

  auto stage = [&](int buf, int kko) {
    // ---- H tile: Hs[m][k] = relu(q[m] . W1[kk+k,0:4] + b1[kk+k]), 2x b128 stores
    {
      const int g0 = kko + hg0;
      const v4f q = qs[hm];
      float bb[16];
      #pragma unroll
      for (int c = 0; c < 4; ++c) {
        v4f bl = *(const v4f*)(b1 + g0 + c * 4);
        bb[c * 4 + 0] = bl.x; bb[c * 4 + 1] = bl.y;
        bb[c * 4 + 2] = bl.z; bb[c * 4 + 3] = bl.w;
      }
      v8h hv[2];
      #pragma unroll
      for (int i = 0; i < 16; ++i) {
        v4f a = ((const v4f*)a4)[g0 + i];
        float h = fmaf(q.x, a.x, fmaf(q.y, a.y, fmaf(q.z, a.z, fmaf(q.w, a.w, bb[i]))));
        hv[i >> 3][i & 7] = (_Float16)fmaxf(h, 0.0f);
      }
      *(v8h*)&Hs[buf][hm][hg0]     = hv[0];
      *(v8h*)&Hs[buf][hm][hg0 + 8] = hv[1];
    }
    // ---- B tile: Bs[e][k] = W2h[n_base+e][kk..kk+63]  (128B per thread) ----
    const _Float16* src = brow + kko;
    _Float16* dst = &Bs[buf][t][0];
#if HAVE_ASYNC_LDS
    as1_v4i* g1 = (as1_v4i*)(unsigned long long)(const void*)src;
    as3_v4i* l3 = (as3_v4i*)(unsigned)(unsigned long long)(void*)dst;
    // offset operand must be a literal integer constant expression
    __builtin_amdgcn_global_load_async_to_lds_b128(g1, l3, 0, 0);
    __builtin_amdgcn_global_load_async_to_lds_b128(g1, l3, 16, 0);
    __builtin_amdgcn_global_load_async_to_lds_b128(g1, l3, 32, 0);
    __builtin_amdgcn_global_load_async_to_lds_b128(g1, l3, 48, 0);
    __builtin_amdgcn_global_load_async_to_lds_b128(g1, l3, 64, 0);
    __builtin_amdgcn_global_load_async_to_lds_b128(g1, l3, 80, 0);
    __builtin_amdgcn_global_load_async_to_lds_b128(g1, l3, 96, 0);
    __builtin_amdgcn_global_load_async_to_lds_b128(g1, l3, 112, 0);
#else
    #pragma unroll
    for (int c = 0; c < 8; ++c) {
      v8h v = *(const v8h*)(src + c * 8);
      *(v8h*)&dst[c * 8] = v;
    }
    if (kko + BK < FFN) __builtin_prefetch(src + BK, 0, 0);  // global_prefetch_b8
#endif
  };

  stage(0, 0);  // prologue

  for (int kk = 0; kk < FFN; kk += BK) {
    const int buf = (kk >> 6) & 1;
#if HAVE_ASYNC_LDS
    wait_async_zero();        // this wave's async fills of `buf` have landed
#endif
    __syncthreads();          // all waves staged `buf`; prior reads of buf^1 done

    if (kk + BK < FFN) stage(buf ^ 1, kk + BK);  // overlap with WMMAs below

    // ---- 2 k-steps x (2 M x 4 N) = 16 WMMAs ----
    #pragma unroll
    for (int kc = 0; kc < 2; ++kc) {
      const int ko = kc * 32 + frag_k;
      v16h af0 = load_frag(&Hs[buf][wm      + frag_row][ko]);
      v16h af1 = load_frag(&Hs[buf][wm + 16 + frag_row][ko]);
      #pragma unroll
      for (int j = 0; j < 4; ++j) {
        v16h bf = load_frag(&Bs[buf][wn + j * 16 + frag_row][ko]);
        acc[0][j] = __builtin_amdgcn_wmma_f32_16x16x32_f16(
            false, af0, false, bf, (short)0, acc[0][j], false, false);
        acc[1][j] = __builtin_amdgcn_wmma_f32_16x16x32_f16(
            false, af1, false, bf, (short)0, acc[1][j], false, false);
      }
    }
  }

  // ---- epilogue: + b2, store f32 ----
  // C/D layout: VGPR r -> M = r + (lane/16)*8 ; lane%16 -> N
  const int col_in = lane & 15;
  const int row_hi = (lane >> 4) * 8;
  #pragma unroll
  for (int j = 0; j < 4; ++j) {
    const int e = n_base + wn + j * 16 + col_in;
    const float bias = b2[e];
    #pragma unroll
    for (int i = 0; i < 2; ++i) {
      #pragma unroll
      for (int r = 0; r < 8; ++r) {
        const int m = m_base + wm + i * 16 + row_hi + r;
        out[(size_t)m * EMBED + e] = acc[i][j][r] + bias;
      }
    }
  }
}

// ---------------- host launch ----------------

extern "C" void kernel_launch(void* const* d_in, const int* in_sizes, int n_in,
                              void* d_out, int out_size, void* d_ws, size_t ws_size,
                              hipStream_t stream) {
  const float* x  = (const float*)d_in[0];   // [8,4096,2048]
  const float* W1 = (const float*)d_in[1];   // [2048,2048]
  const float* b1 = (const float*)d_in[2];   // [2048]
  const float* W2 = (const float*)d_in[3];   // [2048,2048]
  const float* b2 = (const float*)d_in[4];   // [2048]
  float* out = (float*)d_out;

  // workspace layout
  _Float16* W2h = (_Float16*)d_ws;                                        // 8 MB
  float*    qf  = (float*)((char*)d_ws + (size_t)EMBED * FFN * 2);        // 512 KB
  float*    a4  = (float*)((char*)qf + (size_t)NTOK * 4 * sizeof(float)); // 32 KB

  const int n4 = EMBED * FFN / 4;
  cvt_w2_f16<<<(n4 + 255) / 256, 256, 0, stream>>>(W2, W2h, n4);
  qfeat_kernel<<<(NTOK + 255) / 256, 256, 0, stream>>>(x, qf, NTOK);
  pack_a4<<<(FFN + 255) / 256, 256, 0, stream>>>(W1, a4, FFN);

  dim3 grid(EMBED / BN, NTOK / BM);
  ffn_wmma_kernel<<<grid, 256, 0, stream>>>(qf, a4, b1, W2h, b2, out);
}
